// selfattention_38757784879407
// MI455X (gfx1250) — matrix operand order repaired and checked by reading, not compiled, not run on hardware
//
#include <hip/hip_runtime.h>
#include <hip/hip_bf16.h>

// ---------------------------------------------------------------------------
// Fused non-local self-attention block for MI455X (gfx1250, wave32, WMMA).
// B=16, C=64, H=W=256 (HW=65536). bf16 WMMA, f32 accumulation. Softmaxes are
// folded algebraically so no [B,C,HW] intermediate touches HBM (~800 MB total
// traffic -> ~35us at 23.3 TB/s; ~43 GFLOP of bf16 WMMA).
//
// Round-4: K1 processes 4 pixel chunks per block with the similarity (S) and
// Z accumulators held in registers across chunks -> 4x fewer f32 atomics on
// the hot 256KB S buffer and 4x amortized weight staging. GEMM operands all
// load as ds_load_b128 pairs; epilogues are lane-contiguous 16B LDS stores /
// float4 global b128 ops (established rounds 2-3).
// ---------------------------------------------------------------------------

typedef __attribute__((ext_vector_type(16))) __bf16 v16bf;
typedef __attribute__((ext_vector_type(8)))  __bf16 v8bf;
typedef __attribute__((ext_vector_type(4)))  __bf16 v4bf;
typedef __attribute__((ext_vector_type(8)))  float  v8f;

constexpr int CC  = 64;        // channels
constexpr int HWN = 256 * 256; // H*W
constexpr int TN  = 256;       // pixels per tile
constexpr int NCH = 4;         // chunks per K1 block (register-resident S acc)
constexpr int NB  = 16;        // batch

#define WMMA_BF16(A, B, Cacc) \
    __builtin_amdgcn_wmma_f32_16x16x32_bf16(false, (A), false, (B), (short)0, (Cacc), false, false)

// ---- fragment loaders (contiguous k = 16h+s, same for A and B, wave32) ------------------
// A (16-bit 16x32 MxK): lane<16 -> M=lane, K=k0..k0+15 (lane>=16: +16).
__device__ __forceinline__ v16bf ldA16(const __bf16* p) {
    v8bf lo = *reinterpret_cast<const v8bf*>(p);
    v8bf hi = *reinterpret_cast<const v8bf*>(p + 8);
    return __builtin_shufflevector(lo, hi, 0, 1, 2, 3, 4, 5, 6, 7,
                                           8, 9, 10, 11, 12, 13, 14, 15);
}
// B (16-bit 32x16 KxN): lane<16 -> N=lane, K=k0..k0+15 (lane>=16: +16).
__device__ __forceinline__ v16bf ldB16(const __bf16* p) {
    return *reinterpret_cast<const v16bf*>(p);
}
// C/D (f32 16x16): element i -> row = i + 8*hl, col = lane%16.

__device__ __forceinline__ float lrelu(float v) { return v >= 0.0f ? v : 0.1f * v; }

// one 16x16 tile of S[c,d] += sum_n ps[c,n]*es[d,n] over a TN-pixel chunk
__device__ __forceinline__ v8f s_tile(const __bf16* ps, const __bf16* es,
                                      int mt, int nt, int hl, int l16, v8f acc) {
    #pragma unroll
    for (int k0 = 0; k0 < TN; k0 += 32) {
        v16bf af  = ldA16(&ps[(mt + l16) * TN + k0 + hl * 16]);
        v16bf bfr = ldB16(&es[(nt + l16) * TN + k0 + hl * 16]);
        acc = WMMA_BF16(af, bfr, acc);
    }
    return acc;
}

// ---------------------------------------------------------------------------
__global__ void k0_zero(float* __restrict__ p, int n) {
    int i = blockIdx.x * 256 + threadIdx.x;
    if (i < n) p[i] = 0.0f;
}

// ---------------------------------------------------------------------------
// K1: theta & phi projections for NCH 64 x TN tiles; accumulates
//     S[b,c,d] += sum_n phi[c,n]*exp(theta[d,n]) and Z[b,d] += sum_n exp
//     in registers, committing one atomic per element at the end.
// ---------------------------------------------------------------------------
__global__ __launch_bounds__(256)
void k1_proj_sim(const float* __restrict__ x,
                 const float* __restrict__ thw, const float* __restrict__ thb,
                 const float* __restrict__ phw, const float* __restrict__ phb,
                 float* __restrict__ Sg, float* __restrict__ Zg)
{
    __shared__ alignas(32) __bf16 wth[CC * CC];   // theta W, row-major [o][c]
    __shared__ alignas(32) __bf16 wph[CC * CC];   // phi   W, row-major [o][c]
    __shared__ alignas(32) __bf16 xsT[TN * CC];   // x tile TRANSPOSED [n][c]
    __shared__ alignas(32) __bf16 es [CC * TN];   // exp(theta) [d][n]
    __shared__ alignas(32) __bf16 ps [CC * TN];   // phi        [c][n]
    __shared__ float bth[CC], bph[CC];

    const int b    = blockIdx.y;
    const int tid  = threadIdx.x;
    const int lane = tid & 31;
    const int wv   = tid >> 5;       // 8 waves
    const int hl   = lane >> 4;
    const int l16  = lane & 15;

    // ---- stage weights once per block (float4 read -> v4bf packed store) --
    for (int i = tid; i < (CC * CC) / 4; i += 256) {
        float4 t = reinterpret_cast<const float4*>(thw)[i];
        float4 p = reinterpret_cast<const float4*>(phw)[i];
        v4bf tv = {(__bf16)t.x, (__bf16)t.y, (__bf16)t.z, (__bf16)t.w};
        v4bf pv = {(__bf16)p.x, (__bf16)p.y, (__bf16)p.z, (__bf16)p.w};
        *reinterpret_cast<v4bf*>(&wth[i * 4]) = tv;
        *reinterpret_cast<v4bf*>(&wph[i * 4]) = pv;
    }
    if (tid < CC) { bth[tid] = thb[tid]; bph[tid] = phb[tid]; }

    // wave's two S tiles: same c-tile, d-tiles nt0 and nt0+32
    const int mt0 = (wv & 3) << 4;
    const int nt0 = (wv >> 2) << 4;
    v8f accS0 = {}, accS1 = {};
    float zacc = 0.0f;

    for (int ch = 0; ch < NCH; ++ch) {
        const int col0 = (blockIdx.x * NCH + ch) * TN;
        const float* xb = x + (size_t)b * CC * HWN + col0;

        // protects xsT/es/ps against readers from the previous chunk
        __syncthreads();

        // ---- stage x tile transposed: 4 coalesced rows -> one v4bf store --
        for (int i = tid; i < (CC / 4) * TN; i += 256) {
            int n  = i & (TN - 1);
            int c4 = (i >> 8) << 2;
            v4bf v = {(__bf16)xb[(size_t)(c4 + 0) * HWN + n],
                      (__bf16)xb[(size_t)(c4 + 1) * HWN + n],
                      (__bf16)xb[(size_t)(c4 + 2) * HWN + n],
                      (__bf16)xb[(size_t)(c4 + 3) * HWN + n]};
            *reinterpret_cast<v4bf*>(&xsT[n * CC + c4]) = v;
        }
        __syncthreads();

        // ---- GEMM1 (swapped): D[n][m] = sum_c x^T[n][c] * W^T[c][m] ------
        // Epilogue: channel uniform per lane, 8 contiguous pixels -> 16B store.
        for (int r = 0; r < 2; ++r) {
            const int pt = (wv + (r << 3)) << 4;        // pixel tile (M)
            const __bf16* ar = &xsT[(pt + l16) * CC + hl * 16];
            v16bf a0 = ldA16(ar);                       // K(c) 0..31
            v16bf a1 = ldA16(ar + 32);                  // K(c) 32..63
            #pragma unroll
            for (int ct4 = 0; ct4 < 4; ++ct4) {
                const int ct = ct4 << 4;                // channel tile (N)
                const __bf16* tr = &wth[(ct + l16) * CC + hl * 16];
                const __bf16* pr = &wph[(ct + l16) * CC + hl * 16];
                v8f accT = {};
                accT = WMMA_BF16(a0, ldB16(tr),      accT);
                accT = WMMA_BF16(a1, ldB16(tr + 32), accT);
                v8f accP = {};
                accP = WMMA_BF16(a0, ldB16(pr),      accP);
                accP = WMMA_BF16(a1, ldB16(pr + 32), accP);
                const int m  = ct + l16;    // output channel (uniform/lane)
                const int nb = pt + hl * 8; // first pixel of this lane's run
                const float bt = bth[m], bp = bph[m];
                v8bf ev, pv;
                #pragma unroll
                for (int i = 0; i < 8; ++i) {
                    ev[i] = (__bf16)__expf(accT[i] + bt);
                    pv[i] = (__bf16)(accP[i] + bp);
                }
                *reinterpret_cast<v8bf*>(&es[m * TN + nb]) = ev;
                *reinterpret_cast<v8bf*>(&ps[m * TN + nb]) = pv;
            }
        }
        __syncthreads();

        // ---- Z partial sums (register-resident across chunks) ----
        if (tid < CC) {
            const __bf16* row = &es[tid * TN];
            for (int n = 0; n < TN; n += 8) {
                v8bf v = *reinterpret_cast<const v8bf*>(&row[n]);
                #pragma unroll
                for (int j = 0; j < 8; ++j) zacc += (float)v[j];
            }
        }

        // ---- GEMM2: accumulate this chunk into the wave's two S tiles ----
        accS0 = s_tile(ps, es, mt0, nt0,      hl, l16, accS0);
        accS1 = s_tile(ps, es, mt0, nt0 + 32, hl, l16, accS1);
    }

    // ---- commit: one atomic per element for the whole 4-chunk span ----
    if (tid < CC) atomicAdd(&Zg[b * CC + tid], zacc);
    #pragma unroll
    for (int i = 0; i < 8; ++i) {
        const int c = mt0 + hl * 8 + i;
        atomicAdd(&Sg[(b << 12) + c * CC + nt0 + l16],      accS0[i]);
        atomicAdd(&Sg[(b << 12) + c * CC + nt0 + 32 + l16], accS1[i]);
    }
}

// ---------------------------------------------------------------------------
// K2: sim[b,c,d] = S[b,c,d] / Z[b,d], stored bf16 for the K3 WMMA B-operand.
// ---------------------------------------------------------------------------
__global__ void k2_norm(const float* __restrict__ Sg, const float* __restrict__ Zg,
                        __bf16* __restrict__ simB)
{
    int i = blockIdx.x * 256 + threadIdx.x;           // NB*CC*CC = 65536
    int b = i >> 12, d = i & 63;
    simB[i] = (__bf16)(Sg[i] / Zg[b * CC + d]);
}

// ---------------------------------------------------------------------------
// K3: g projection + channel softmax + attention = sim @ g_y, residual+lrelu
// ---------------------------------------------------------------------------
__global__ __launch_bounds__(256)
void k3_att(const float* __restrict__ x,
            const float* __restrict__ gw, const float* __restrict__ gb,
            const __bf16* __restrict__ simB, float* __restrict__ out)
{
    __shared__ alignas(32) __bf16 wg [CC * CC];   // [o][c]
    __shared__ alignas(32) __bf16 sim[CC * CC];   // [c][d]
    __shared__ alignas(32) __bf16 xsT[TN * CC];   // x tile transposed [n][c]
    __shared__ alignas(32) __bf16 egT[TN * CC];   // exp(g) TRANSPOSED [n][d]
    __shared__ float gsum[TN];                    // reciprocal of channel sums
    __shared__ float bg[CC];

    const int b    = blockIdx.y;
    const int col0 = blockIdx.x * TN;
    const int tid  = threadIdx.x;
    const int lane = tid & 31;
    const int wv   = tid >> 5;
    const int hl   = lane >> 4;
    const int l16  = lane & 15;

    for (int i = tid; i < (CC * CC) / 4; i += 256) {
        float4 g = reinterpret_cast<const float4*>(gw)[i];
        v4bf gv = {(__bf16)g.x, (__bf16)g.y, (__bf16)g.z, (__bf16)g.w};
        *reinterpret_cast<v4bf*>(&wg[i * 4]) = gv;
    }
    for (int i = tid; i < (CC * CC) / 8; i += 256) {   // 16B copies of sim
        reinterpret_cast<uint4*>(sim)[i] =
            reinterpret_cast<const uint4*>(simB + ((size_t)b << 12))[i];
    }
    if (tid < CC) bg[tid] = gb[tid];
    const float* xb = x + (size_t)b * CC * HWN + col0;
    for (int i = tid; i < (CC / 4) * TN; i += 256) {
        int n  = i & (TN - 1);
        int c4 = (i >> 8) << 2;
        v4bf v = {(__bf16)xb[(size_t)(c4 + 0) * HWN + n],
                  (__bf16)xb[(size_t)(c4 + 1) * HWN + n],
                  (__bf16)xb[(size_t)(c4 + 2) * HWN + n],
                  (__bf16)xb[(size_t)(c4 + 3) * HWN + n]};
        *reinterpret_cast<v4bf*>(&xsT[n * CC + c4]) = v;
    }
    __syncthreads();

    // ---- GEMM1: g = Wg @ x (A=W, B=x^T); D[d][n] -> egT[n][d] 16B stores --
    for (int r = 0; r < 2; ++r) {
        const int nt = (wv + (r << 3)) << 4;
        const __bf16* xr = &xsT[(nt + l16) * CC + hl * 16];
        v16bf b0 = ldB16(xr);
        v16bf b1 = ldB16(xr + 32);
        #pragma unroll
        for (int mt4 = 0; mt4 < 4; ++mt4) {
            const int mt = mt4 << 4;
            const __bf16* wr = &wg[(mt + l16) * CC + hl * 16];
            v8f acc = {};
            acc = WMMA_BF16(ldA16(wr),      b0, acc);
            acc = WMMA_BF16(ldA16(wr + 32), b1, acc);
            float4 bg0 = *reinterpret_cast<const float4*>(&bg[mt + hl * 8]);
            float4 bg1 = *reinterpret_cast<const float4*>(&bg[mt + hl * 8 + 4]);
            v8bf ev;
            ev[0] = (__bf16)__expf(acc[0] + bg0.x);
            ev[1] = (__bf16)__expf(acc[1] + bg0.y);
            ev[2] = (__bf16)__expf(acc[2] + bg0.z);
            ev[3] = (__bf16)__expf(acc[3] + bg0.w);
            ev[4] = (__bf16)__expf(acc[4] + bg1.x);
            ev[5] = (__bf16)__expf(acc[5] + bg1.y);
            ev[6] = (__bf16)__expf(acc[6] + bg1.z);
            ev[7] = (__bf16)__expf(acc[7] + bg1.w);
            *reinterpret_cast<v8bf*>(&egT[(nt + l16) * CC + mt + hl * 8]) = ev;
        }
    }
    __syncthreads();

    // ---- channel-softmax reciprocal denominator per pixel ----
    {
        float s = 0.0f;
        const __bf16* row = &egT[tid * CC];
        #pragma unroll
        for (int d = 0; d < CC; d += 8) {
            v8bf v = *reinterpret_cast<const v8bf*>(&row[d]);
            #pragma unroll
            for (int j = 0; j < 8; ++j) s += (float)v[j];
        }
        gsum[tid] = 1.0f / s;
    }
    __syncthreads();

    // ---- GEMM2 (swapped): D[n][c] = sum_d egT[n][d] * sim^T[d][c] ---------
    // Epilogue per lane: 8 consecutive pixels of one channel -> float4 pairs.
    for (int r = 0; r < 2; ++r) {
        const int pt = (wv + (r << 3)) << 4;            // pixel tile (M)
        const __bf16* ar = &egT[(pt + l16) * CC + hl * 16];
        v16bf a0 = ldA16(ar);
        v16bf a1 = ldA16(ar + 32);
        #pragma unroll
        for (int ct4 = 0; ct4 < 4; ++ct4) {
            const int ct = ct4 << 4;                    // channel tile (N)
            const __bf16* sr = &sim[(ct + l16) * CC + hl * 16];
            v8f acc = {};
            acc = WMMA_BF16(a0, ldB16(sr),      acc);
            acc = WMMA_BF16(a1, ldB16(sr + 32), acc);
            const int c  = ct + l16;      // channel (uniform per lane)
            const int nb = pt + hl * 8;   // first pixel of this lane's run
            const float* xr = xb + (size_t)c * HWN + nb;
            float4 x0 = *reinterpret_cast<const float4*>(xr);
            float4 x1 = *reinterpret_cast<const float4*>(xr + 4);
            float4 z0 = *reinterpret_cast<const float4*>(&gsum[nb]);
            float4 z1 = *reinterpret_cast<const float4*>(&gsum[nb + 4]);
            float4 o0, o1;
            o0.x = lrelu(x0.x + acc[0] * z0.x);
            o0.y = lrelu(x0.y + acc[1] * z0.y);
            o0.z = lrelu(x0.z + acc[2] * z0.z);
            o0.w = lrelu(x0.w + acc[3] * z0.w);
            o1.x = lrelu(x1.x + acc[4] * z1.x);
            o1.y = lrelu(x1.y + acc[5] * z1.y);
            o1.z = lrelu(x1.z + acc[6] * z1.z);
            o1.w = lrelu(x1.w + acc[7] * z1.w);
            float* orow = out + ((size_t)b * CC + c) * HWN + col0 + nb;
            *reinterpret_cast<float4*>(orow)     = o0;
            *reinterpret_cast<float4*>(orow + 4) = o1;
        }
    }
}

// ---------------------------------------------------------------------------
extern "C" void kernel_launch(void* const* d_in, const int* in_sizes, int n_in,
                              void* d_out, int out_size, void* d_ws, size_t ws_size,
                              hipStream_t stream)
{
    const float* x  = (const float*)d_in[0];
    const float* gw = (const float*)d_in[1];
    const float* gb = (const float*)d_in[2];
    const float* tw = (const float*)d_in[3];
    const float* tb = (const float*)d_in[4];
    const float* pw = (const float*)d_in[5];
    const float* pb = (const float*)d_in[6];
    float* out = (float*)d_out;

    // workspace carve-out: S (f32), Z (f32), sim (bf16) -- ~0.4 MB total
    float*  Sg   = (float*)d_ws;                 // NB*CC*CC
    float*  Zg   = Sg + NB * CC * CC;            // NB*CC
    __bf16* simB = (__bf16*)(Zg + NB * CC);      // NB*CC*CC

    const int zn = NB * CC * CC + NB * CC;       // S and Z are contiguous
    k0_zero<<<dim3((zn + 255) / 256), dim3(256), 0, stream>>>(Sg, zn);

    dim3 gridK1(HWN / (TN * NCH), NB);           // 64 x 16
    k1_proj_sim<<<gridK1, dim3(256), 0, stream>>>(x, tw, tb, pw, pb, Sg, Zg);
    k2_norm<<<dim3((NB * CC * CC) / 256), dim3(256), 0, stream>>>(Sg, Zg, simB);
    dim3 gridK3(HWN / TN, NB);                   // 256 x 16
    k3_att<<<gridK3, dim3(256), 0, stream>>>(x, gw, gb, simB, out);
}
